// GroupwiseProjection_60215441490419
// MI455X (gfx1250) — compile-verified, matrix operand order checked by reading, not run on hardware
//
#include <hip/hip_runtime.h>

typedef __attribute__((ext_vector_type(16))) __bf16 v16bf;
typedef __attribute__((ext_vector_type(8)))  __bf16 v8bf;
typedef __attribute__((ext_vector_type(4)))  __bf16 v4bf;
typedef __attribute__((ext_vector_type(8)))  float  v8f;
typedef __attribute__((ext_vector_type(4)))  float  v4f;
typedef __attribute__((ext_vector_type(4)))  int    v4i;

#define NUM_GROUPS 8
#define D_IN   512
#define D_OUT  512
#define T_TOTAL 4096
#define BATCH  16

#define TM 64
#define TN 128
#define TK 32                 // one WMMA k-step per pipeline stage
#define LDS_STRIDE 40         // 32 bf16 + 8 pad -> 80B rows (16B aligned, bank-spread)

#ifndef __has_builtin
#define __has_builtin(x) 0
#endif
#if __has_builtin(__builtin_amdgcn_global_load_async_to_lds_b128)
#define USE_ASYNC_LDS 1
#else
#define USE_ASYNC_LDS 0
#endif

// exact parameter types leaked by the round-2 diagnostic: int4* in AS1/AS3
typedef __attribute__((address_space(1))) v4i GV4;
typedef __attribute__((address_space(3))) v4i LV4;

__device__ __forceinline__ void wait_async0()
{
#if USE_ASYNC_LDS
# if __has_builtin(__builtin_amdgcn_s_wait_asynccnt)
    __builtin_amdgcn_s_wait_asynccnt(0);
# else
    asm volatile("s_wait_asynccnt 0" ::: "memory");
# endif
#endif
}

// ---------------------------------------------------------------------------
// Pre-pass: split fp32 W into bf16 hi + bf16 residual (lo) planes.
// ---------------------------------------------------------------------------
__global__ void __launch_bounds__(256)
wsplit_kernel(const float* __restrict__ W,
              __bf16* __restrict__ hi, __bf16* __restrict__ lo, int n)
{
    int i = (blockIdx.x * 256 + threadIdx.x) * 4;
    if (i + 3 < n) {
        v4f w = *(const v4f*)(W + i);
        v4bf h, l;
        #pragma unroll
        for (int j = 0; j < 4; ++j) {
            __bf16 hj = (__bf16)w[j];
            h[j] = hj;
            l[j] = (__bf16)(w[j] - (float)hj);
        }
        *(v4bf*)(hi + i) = h;
        *(v4bf*)(lo + i) = l;
    }
}

// ---------------------------------------------------------------------------
// Fragment load (CDNA5 16-bit A/B WMMA layout): lane l covers row (l&15);
// elements 0..7 -> k = koff..koff+7, 8..15 -> koff+16.., koff += 8 for
// lanes >= 16.  Two 16-byte DS loads.
// ---------------------------------------------------------------------------
__device__ __forceinline__ v16bf frag_ld(const __bf16* base, int row, int lane)
{
    int koff = (lane & 16) ? 8 : 0;
    const __bf16* p = base + row * LDS_STRIDE + koff;
    v8bf a = *(const v8bf*)(p);
    v8bf b = *(const v8bf*)(p + 16);
    v16bf r;
    #pragma unroll
    for (int i = 0; i < 8; ++i) { r[i] = a[i]; r[i + 8] = b[i]; }
    return r;
}

// ---------------------------------------------------------------------------
// Main GEMM: block tile 64x128, 8 waves (4M x 2N), double-buffered K pipeline,
// split-bf16 (hi/lo) emulated fp32: acc += al*bh + ah*bl + ah*bh.
// ---------------------------------------------------------------------------
template <bool PRECONV>
__global__ void __launch_bounds__(256)
gproj_kernel(const float* __restrict__ x,
             const float* __restrict__ W,
             const float* __restrict__ bias,
             const __bf16* __restrict__ Whi,
             const __bf16* __restrict__ Wlo,
             float* __restrict__ out)
{
    __shared__ __bf16 sA[2][2][TM * LDS_STRIDE];   // [stage][hi/lo]
    __shared__ __bf16 sB[2][2][TN * LDS_STRIDE];

    const int tid  = threadIdx.x;
    const int lane = tid & 31;
    const int wave = tid >> 5;
    const int wm   = wave >> 1;   // 0..3
    const int wn   = wave & 1;    // 0..1

    const int row_base = blockIdx.x * TM;          // flattened (b*T + t)
    const int n0       = blockIdx.y * TN;

    // group lookup (uniform per block; all group starts are multiples of 64)
    const int t0 = row_base & (T_TOTAL - 1);
    const int starts[NUM_GROUPS + 1] = {0, 256, 768, 1536, 1920, 2560, 3072, 3648, 4096};
    int g = 0;
    #pragma unroll
    for (int i = 1; i < NUM_GROUPS; ++i) g += (t0 >= starts[i]) ? 1 : 0;

    const size_t wbase = (size_t)g * D_OUT * D_IN + (size_t)n0 * D_IN;
    const float*  wg = W + wbase;
    const __bf16* wh = PRECONV ? (Whi + wbase) : nullptr;
    const __bf16* wl = PRECONV ? (Wlo + wbase) : nullptr;

    // ---- staging helpers (lambdas keep everything inlined) -----------------
    auto loadA = [&](int kc, v4f xv[2]) {
        #pragma unroll
        for (int p = 0; p < 2; ++p) {
            int idx = tid + p * 256;               // 64 rows x 8 float4
            int r   = idx >> 3;
            int c4  = (idx & 7) * 4;
            xv[p] = *(const v4f*)(x + (size_t)(row_base + r) * D_IN + kc + c4);
        }
    };
    auto storeA = [&](const v4f xv[2], int st) {
        #pragma unroll
        for (int p = 0; p < 2; ++p) {
            int idx = tid + p * 256;
            int r   = idx >> 3;
            int c4  = (idx & 7) * 4;
            v4bf h, l;
            #pragma unroll
            for (int j = 0; j < 4; ++j) {
                __bf16 hj = (__bf16)xv[p][j];
                h[j] = hj;
                l[j] = (__bf16)(xv[p][j] - (float)hj);
            }
            *(v4bf*)(&sA[st][0][r * LDS_STRIDE + c4]) = h;
            *(v4bf*)(&sA[st][1][r * LDS_STRIDE + c4]) = l;
        }
    };
    auto stageB = [&](int kc, int st) {
        if (PRECONV) {
#if USE_ASYNC_LDS
            #pragma unroll
            for (int p = 0; p < 2; ++p) {
                int idx = tid + p * 256;           // 128 rows x 4 b128 chunks
                int r   = idx >> 2;
                int c8  = (idx & 3) * 8;
                size_t go = (size_t)r * D_IN + kc + c8;
                int    lo = r * LDS_STRIDE + c8;
                __builtin_amdgcn_global_load_async_to_lds_b128(
                    (GV4*)(wh + go), (LV4*)(&sB[st][0][lo]), 0, 0);
                __builtin_amdgcn_global_load_async_to_lds_b128(
                    (GV4*)(wl + go), (LV4*)(&sB[st][1][lo]), 0, 0);
            }
#else
            #pragma unroll
            for (int p = 0; p < 2; ++p) {
                int idx = tid + p * 256;
                int r   = idx >> 2;
                int c8  = (idx & 3) * 8;
                size_t go = (size_t)r * D_IN + kc + c8;
                int    lo = r * LDS_STRIDE + c8;
                *(v8bf*)(&sB[st][0][lo]) = *(const v8bf*)(wh + go);
                *(v8bf*)(&sB[st][1][lo]) = *(const v8bf*)(wl + go);
            }
#endif
        } else {
            #pragma unroll
            for (int p = 0; p < 4; ++p) {
                int idx = tid + p * 256;           // 128 rows x 8 float4
                int r   = idx >> 3;
                int c4  = (idx & 7) * 4;
                v4f wv = *(const v4f*)(wg + (size_t)r * D_IN + kc + c4);
                v4bf h, l;
                #pragma unroll
                for (int j = 0; j < 4; ++j) {
                    __bf16 hj = (__bf16)wv[j];
                    h[j] = hj;
                    l[j] = (__bf16)(wv[j] - (float)hj);
                }
                *(v4bf*)(&sB[st][0][r * LDS_STRIDE + c4]) = h;
                *(v4bf*)(&sB[st][1][r * LDS_STRIDE + c4]) = l;
            }
        }
    };

    v8f acc[4];
    #pragma unroll
    for (int nt = 0; nt < 4; ++nt)
        #pragma unroll
        for (int r = 0; r < 8; ++r) acc[nt][r] = 0.0f;

    // ---- prologue: stage chunk 0 into buffer 0 -----------------------------
    {
        v4f xv[2];
        loadA(0, xv);
        stageB(0, 0);
        storeA(xv, 0);
        wait_async0();
        __syncthreads();
    }

    // ---- pipelined K loop --------------------------------------------------
    const int arow = wm * 16 + (lane & 15);
    #pragma unroll 2
    for (int kc = 0; kc < D_IN; kc += TK) {
        const int cur  = (kc >> 5) & 1;
        const int nxt  = cur ^ 1;
        const bool more = (kc + TK) < D_IN;

        v4f xv[2];
        if (more) {
            loadA(kc + TK, xv);       // global loads in flight during WMMAs
            stageB(kc + TK, nxt);     // async DMA into next-stage LDS
        }

        // compute on current stage: 12 WMMAs
        v16bf ah = frag_ld(sA[cur][0], arow, lane);
        v16bf al = frag_ld(sA[cur][1], arow, lane);
        v16bf bh[4], bl[4];
        #pragma unroll
        for (int nt = 0; nt < 4; ++nt) {
            int brow = wn * 64 + nt * 16 + (lane & 15);
            bh[nt] = frag_ld(sB[cur][0], brow, lane);
            bl[nt] = frag_ld(sB[cur][1], brow, lane);
        }
        #pragma unroll
        for (int nt = 0; nt < 4; ++nt)
            acc[nt] = __builtin_amdgcn_wmma_f32_16x16x32_bf16(
                false, al, false, bh[nt], (short)0, acc[nt], false, false);
        #pragma unroll
        for (int nt = 0; nt < 4; ++nt)
            acc[nt] = __builtin_amdgcn_wmma_f32_16x16x32_bf16(
                false, ah, false, bl[nt], (short)0, acc[nt], false, false);
        #pragma unroll
        for (int nt = 0; nt < 4; ++nt)
            acc[nt] = __builtin_amdgcn_wmma_f32_16x16x32_bf16(
                false, ah, false, bh[nt], (short)0, acc[nt], false, false);

        if (more) storeA(xv, nxt);    // f32->bf16 split in WMMA shadow
        wait_async0();
        __syncthreads();
    }

    // ---- epilogue: bias + store (C/D layout: m = r + 8*(lane>=16), n = lane&15)
    const int mbase = row_base + wm * 16 + ((lane & 16) ? 8 : 0);
    #pragma unroll
    for (int nt = 0; nt < 4; ++nt) {
        int col  = n0 + wn * 64 + nt * 16 + (lane & 15);
        float bv = bias[g * D_OUT + col];
        #pragma unroll
        for (int r = 0; r < 8; ++r)
            out[(size_t)(mbase + r) * D_OUT + col] = acc[nt][r] + bv;
    }
}

// ---------------------------------------------------------------------------
extern "C" void kernel_launch(void* const* d_in, const int* in_sizes, int n_in,
                              void* d_out, int out_size, void* d_ws, size_t ws_size,
                              hipStream_t stream)
{
    (void)in_sizes; (void)n_in; (void)out_size;
    const float* x    = (const float*)d_in[0];
    const float* W    = (const float*)d_in[1];
    const float* bias = (const float*)d_in[2];
    float* out        = (float*)d_out;

    const size_t welems = (size_t)NUM_GROUPS * D_OUT * D_IN;   // 2,097,152
    dim3 grid(BATCH * T_TOTAL / TM, D_OUT / TN);               // (1024, 4)

    if (ws_size >= welems * 2 * sizeof(__bf16)) {              // 8 MB scratch
        __bf16* whi = (__bf16*)d_ws;
        __bf16* wlo = whi + welems;
        wsplit_kernel<<<(int)(welems / 1024), 256, 0, stream>>>(W, whi, wlo, (int)welems);
        gproj_kernel<true><<<grid, 256, 0, stream>>>(x, W, bias, whi, wlo, out);
    } else {
        gproj_kernel<false><<<grid, 256, 0, stream>>>(x, W, bias, nullptr, nullptr, out);
    }
}
